// AdaptiveDecayMemory_71347996721398
// MI455X (gfx1250) — matrix operand order repaired and checked by reading, not compiled, not run on hardware
//
#include <hip/hip_runtime.h>
#include <cstdint>

typedef __attribute__((ext_vector_type(16))) __bf16 v16bf;
typedef __attribute__((ext_vector_type(8)))  __bf16 v8bf;
typedef __attribute__((ext_vector_type(8)))  float  v8f;

#define DIMN 1024
#define BSZ  4
#define TLEN 2048
#define NTOK (BSZ * TLEN)

// ---------------- fp32 -> bf16 conversion ----------------
__global__ void cvt_bf16_kernel(const float* __restrict__ in, __bf16* __restrict__ out, int n) {
  int i = blockIdx.x * blockDim.x + threadIdx.x;
  if (i < n) out[i] = (__bf16)in[i];
}

// ---------------- per-token decay: log(sigmoid(x . Wd + bd) + 1e-8) ----------------
__global__ __launch_bounds__(256) void decay_kernel(const float* __restrict__ x,
                                                    const float* __restrict__ wd,
                                                    const float* __restrict__ bd,
                                                    float* __restrict__ logdecay) {
  const int wave = threadIdx.x >> 5;
  const int lane = threadIdx.x & 31;
  const int tok  = blockIdx.x * 8 + wave;          // grid = NTOK/8
  const float* xr = x + (size_t)tok * DIMN;
  float s = 0.f;
  for (int d = lane; d < DIMN; d += 32) s += xr[d] * wd[d];
#pragma unroll
  for (int off = 16; off > 0; off >>= 1) s += __shfl_xor(s, off, 32);
  if (lane == 0) {
    float z   = s + bd[0];
    float dec = 1.f / (1.f + __expf(-z));
    logdecay[tok] = __logf(dec + 1e-8f);
  }
}

// ---- CDNA5 async global->LDS copy (16B per lane), tracked by ASYNCcnt ----
__device__ __forceinline__ void async_cp16(void* lds_ptr, const void* gptr) {
  // flat LDS addresses carry the wave-relative LDS offset in bits [31:0]
  unsigned l = (unsigned)(uintptr_t)lds_ptr;
  asm volatile("global_load_async_to_lds_b128 %0, %1, off"
               :: "v"(l), "v"(gptr) : "memory");
}

// ---------------- generic 128x128 WMMA bf16 GEMM (NT), async double-buffered ------
// C[M,N] = A[M,K] * B^T, A row-major [M,K], B row-major [N,K] (K contiguous both).
// EPI = 0: store bf16
// EPI = 1: scores epilogue: val *= (1/32) * decay_weight(i,j); store bf16
// EPI = 2: val *= outscale[0]; store fp32
// EPI = 3: store bf16 TRANSPOSED per batch: Ct[b][col][t], t = row % TLEN
template <int EPI>
__global__ __launch_bounds__(256)
void gemm_wmma_kernel(const __bf16* __restrict__ A, const __bf16* __restrict__ Bm,
                      void* __restrict__ Cv, int K, int lda, int ldb, int ldc,
                      long long strideA, long long strideB, long long strideC,
                      const float* __restrict__ logdecay, const float* __restrict__ outscale) {
  __shared__ __align__(16) __bf16 As[2][128][40];
  __shared__ __align__(16) __bf16 Bs[2][128][40];

  const int b  = blockIdx.z;
  const int i0 = blockIdx.y * 128;
  const int j0 = blockIdx.x * 128;
  const int tid  = threadIdx.x;
  const int wave = tid >> 5, lane = tid & 31;
  const int wm = (wave & 3) * 32;        // wave row offset within tile
  const int wn = (wave >> 2) * 64;       // wave col offset within tile
  const int lr = lane & 15, lh = lane >> 4;

  const __bf16* Ab = A  + (size_t)b * strideA;
  const __bf16* Bb = Bm + (size_t)b * strideB;

  if constexpr (EPI == 1) {
    // tiles strictly below the diagonal: weight==0 everywhere -> store zeros, skip GEMM
    if (j0 + 127 <= i0) {
      __bf16* Cb = (__bf16*)Cv + (size_t)b * strideC;
      for (int e = tid; e < 128 * 128; e += 256) {
        int r = e >> 7, c = e & 127;
        Cb[(size_t)(i0 + r) * ldc + (j0 + c)] = (__bf16)0.f;
      }
      return;
    }
  }

  v8f acc[2][4];
#pragma unroll
  for (int m = 0; m < 2; m++)
#pragma unroll
    for (int t = 0; t < 4; t++)
#pragma unroll
      for (int e = 0; e < 8; e++) acc[m][t][e] = 0.f;

  const int arow = tid >> 1;             // 0..127
  const int ac0  = (tid & 1) * 16;       // 0 or 16
  const __bf16* aBase = Ab + (size_t)(i0 + arow) * lda + ac0;
  const __bf16* bBase = Bb + (size_t)(j0 + arow) * ldb + ac0;

  auto issue = [&](int buf, int k0) {
    async_cp16(&As[buf][arow][ac0],     aBase + k0);
    async_cp16(&As[buf][arow][ac0 + 8], aBase + k0 + 8);
    async_cp16(&Bs[buf][arow][ac0],     bBase + k0);
    async_cp16(&Bs[buf][arow][ac0 + 8], bBase + k0 + 8);
  };

  issue(0, 0);                           // prologue: tile 0 in flight (4 async ops/thread)
  for (int k0 = 0, it = 0; k0 < K; k0 += 32, ++it) {
    const int cur = it & 1;
    if (k0 + 32 < K) {
      issue(cur ^ 1, k0 + 32);           // prefetch next tile into other buffer
      asm volatile("s_wait_asynccnt 0x4" ::: "memory");  // current tile landed (in-order)
    } else {
      asm volatile("s_wait_asynccnt 0x0" ::: "memory");
    }
    __syncthreads();                     // all waves' tiles visible

    // Fragments per ISA 16-bit layouts:
    // A 16x32: lanes<16 hold K{0..7,16..23}, lanes>=16 hold K{8..15,24..31}
    // B 32x16: lane n%16 = column, lanes<16 K{0..7,16..23}, lanes>=16 K{8..15,24..31}
    v16bf fa[2], fb[4];
#pragma unroll
    for (int m = 0; m < 2; m++) {
      const __bf16* r = &As[cur][wm + m * 16 + lr][0];
      v8bf lo = *(const v8bf*)(r + lh * 8);
      v8bf hi = *(const v8bf*)(r + 16 + lh * 8);
      fa[m] = __builtin_shufflevector(lo, hi, 0,1,2,3,4,5,6,7,8,9,10,11,12,13,14,15);
    }
#pragma unroll
    for (int t = 0; t < 4; t++) {
      const __bf16* r = &Bs[cur][wn + t * 16 + lr][0];
      v8bf lo = *(const v8bf*)(r + lh * 8);
      v8bf hi = *(const v8bf*)(r + 16 + lh * 8);
      fb[t] = __builtin_shufflevector(lo, hi, 0,1,2,3,4,5,6,7,8,9,10,11,12,13,14,15);
    }
#pragma unroll
    for (int m = 0; m < 2; m++)
#pragma unroll
      for (int t = 0; t < 4; t++)
        acc[m][t] = __builtin_amdgcn_wmma_f32_16x16x32_bf16(
            false, fa[m], false, fb[t], (short)0, acc[m][t], false, false);

    __syncthreads();                     // all waves done reading buf (cur is now reusable)
  }

  // Epilogue. C/D layout: element v -> M = v + 8*lh, N = lr.
  const float* ldp = nullptr;
  if constexpr (EPI == 1) ldp = logdecay + (size_t)b * TLEN;
#pragma unroll
  for (int m = 0; m < 2; m++) {
#pragma unroll
    for (int t = 0; t < 4; t++) {
      const int col = j0 + wn + t * 16 + lr;
      float ldj = 0.f;
      if constexpr (EPI == 1) ldj = ldp[col];
#pragma unroll
      for (int v = 0; v < 8; v++) {
        const int row = i0 + wm + m * 16 + lh * 8 + v;
        float val = acc[m][t][v];
        if constexpr (EPI == 1) {
          const int diff = col - row;                    // j - i, keep only future
          float w = (diff > 0) ? __expf(ldj * (float)(diff - 1)) : 0.f;
          val *= 0.03125f * w;                           // 1/sqrt(1024) = 1/32
          ((__bf16*)Cv + (size_t)b * strideC)[(size_t)row * ldc + col] = (__bf16)val;
        } else if constexpr (EPI == 2) {
          ((float*)Cv)[(size_t)row * ldc + col] = val * outscale[0];
        } else if constexpr (EPI == 3) {
          const int bb = row >> 11;                      // row / TLEN
          const int tt = row & (TLEN - 1);
          ((__bf16*)Cv)[(size_t)bb * DIMN * TLEN + (size_t)col * TLEN + tt] = (__bf16)val;
        } else {
          ((__bf16*)Cv + (size_t)b * strideC)[(size_t)row * ldc + col] = (__bf16)val;
        }
      }
    }
  }
}

// ---------------- host launch ----------------
extern "C" void kernel_launch(void* const* d_in, const int* in_sizes, int n_in,
                              void* d_out, int out_size, void* d_ws, size_t ws_size,
                              hipStream_t stream) {
  (void)in_sizes; (void)n_in; (void)out_size; (void)ws_size;
  const float* x   = (const float*)d_in[0];
  const float* Wq  = (const float*)d_in[1];
  const float* Wk  = (const float*)d_in[2];
  const float* Wv  = (const float*)d_in[3];
  const float* Wo  = (const float*)d_in[4];
  const float* Wd  = (const float*)d_in[5];
  const float* bd  = (const float*)d_in[6];
  const float* osc = (const float*)d_in[7];

  char* ws = (char*)d_ws;
  size_t off = 0;
  auto alloc = [&](size_t bytes) {
    void* p = ws + off;
    off += (bytes + 255) & ~(size_t)255;
    return p;
  };
  __bf16* xb   = (__bf16*)alloc((size_t)NTOK * DIMN * 2);
  __bf16* wqb  = (__bf16*)alloc((size_t)DIMN * DIMN * 2);
  __bf16* wkb  = (__bf16*)alloc((size_t)DIMN * DIMN * 2);
  __bf16* wvb  = (__bf16*)alloc((size_t)DIMN * DIMN * 2);
  __bf16* wob  = (__bf16*)alloc((size_t)DIMN * DIMN * 2);
  __bf16* qb   = (__bf16*)alloc((size_t)NTOK * DIMN * 2);
  __bf16* kb   = (__bf16*)alloc((size_t)NTOK * DIMN * 2);
  __bf16* vTb  = (__bf16*)alloc((size_t)NTOK * DIMN * 2);   // [B][D][T] transposed
  float*  ldec = (float*) alloc((size_t)NTOK * 4);
  __bf16* P    = (__bf16*)alloc((size_t)BSZ * TLEN * TLEN * 2);
  __bf16* retr = (__bf16*)alloc((size_t)NTOK * DIMN * 2);

  // 1) conversions
  {
    int n = NTOK * DIMN;
    cvt_bf16_kernel<<<dim3((n + 255) / 256), dim3(256), 0, stream>>>(x, xb, n);
    int nw = DIMN * DIMN;
    dim3 g((nw + 255) / 256);
    cvt_bf16_kernel<<<g, dim3(256), 0, stream>>>(Wq, wqb, nw);
    cvt_bf16_kernel<<<g, dim3(256), 0, stream>>>(Wk, wkb, nw);
    cvt_bf16_kernel<<<g, dim3(256), 0, stream>>>(Wv, wvb, nw);
    cvt_bf16_kernel<<<g, dim3(256), 0, stream>>>(Wo, wob, nw);
  }
  // 2) decay logits
  decay_kernel<<<dim3(NTOK / 8), dim3(256), 0, stream>>>(x, Wd, bd, ldec);

  // 3) q,k projections (bf16 out) and v projection stored transposed
  {
    dim3 g(DIMN / 128, NTOK / 128, 1);
    gemm_wmma_kernel<0><<<g, dim3(256), 0, stream>>>(
        xb, wqb, qb, DIMN, DIMN, DIMN, DIMN, 0, 0, 0, nullptr, nullptr);
    gemm_wmma_kernel<0><<<g, dim3(256), 0, stream>>>(
        xb, wkb, kb, DIMN, DIMN, DIMN, DIMN, 0, 0, 0, nullptr, nullptr);
    gemm_wmma_kernel<3><<<g, dim3(256), 0, stream>>>(
        xb, wvb, vTb, DIMN, DIMN, DIMN, TLEN, 0, 0, 0, nullptr, nullptr);
  }
  // 4) P = weight(i,j) * (q k^T / 32)  per batch (decay epilogue, bf16 out)
  {
    dim3 g(TLEN / 128, TLEN / 128, BSZ);
    gemm_wmma_kernel<1><<<g, dim3(256), 0, stream>>>(
        qb, kb, P, DIMN, DIMN, DIMN, TLEN,
        (long long)TLEN * DIMN, (long long)TLEN * DIMN, (long long)TLEN * TLEN,
        ldec, nullptr);
  }
  // 5) retrieved = P * (V^T)^T per batch: NT GEMM against transposed V (bf16 out)
  {
    dim3 g(DIMN / 128, TLEN / 128, BSZ);
    gemm_wmma_kernel<0><<<g, dim3(256), 0, stream>>>(
        P, vTb, retr, TLEN, TLEN, TLEN, DIMN,
        (long long)TLEN * TLEN, (long long)TLEN * DIMN, (long long)TLEN * DIMN,
        nullptr, nullptr);
  }
  // 6) out = retrieved * Wo^T * out_scale  (fp32 out)
  {
    dim3 g(DIMN / 128, NTOK / 128, 1);
    gemm_wmma_kernel<2><<<g, dim3(256), 0, stream>>>(
        retr, wob, d_out, DIMN, DIMN, DIMN, DIMN, 0, 0, 0, nullptr, osc);
  }
}
